// GATTransformer_51762945851655
// MI455X (gfx1250) — matrix operand order repaired and checked by reading, not compiled, not run on hardware
//
#include <hip/hip_runtime.h>
#include <cstdint>

// ---------------- problem constants (match reference) ----------------
#define NNODES 40000
#define NEDGES 320000
#define DIM    128
#define HC     512     // H*C
#define NHEAD  4
#define CH     128
#define MLPD   512
#define DEPTHL 2

typedef __attribute__((ext_vector_type(8)))  float  v8f;
typedef __attribute__((ext_vector_type(16))) __bf16 v16bf;

// ---------------- small device helpers ----------------
__device__ __forceinline__ unsigned short f32_to_bf16(float f) {
  unsigned u = __float_as_uint(f);
  u += 0x7fffu + ((u >> 16) & 1u);      // round to nearest even
  return (unsigned short)(u >> 16);
}

__device__ __forceinline__ float wave_sum(float v) {
  v += __shfl_xor(v, 16);
  v += __shfl_xor(v, 8);
  v += __shfl_xor(v, 4);
  v += __shfl_xor(v, 2);
  v += __shfl_xor(v, 1);
  return v;
}

// order-preserving float -> uint key (for atomicMax on floats)
__device__ __forceinline__ unsigned fkey(float f) {
  unsigned u = __float_as_uint(f);
  return (u & 0x80000000u) ? ~u : (u | 0x80000000u);
}
__device__ __forceinline__ float fkey_inv(unsigned k) {
  unsigned u = (k & 0x80000000u) ? (k & 0x7fffffffu) : ~k;
  return __uint_as_float(u);
}

__device__ __forceinline__ v16bf pack_frag(uint4 lo, uint4 hi) {
  union { uint4 u[2]; v16bf v; } t;
  t.u[0] = lo; t.u[1] = hi;
  return t.v;
}

// ---------------- LayerNorm -> bf16 (wave per row, D=128) ----------------
__global__ void ln_bf16_kernel(const float* __restrict__ x,
                               const float* __restrict__ gm,
                               const float* __restrict__ bt,
                               unsigned short* __restrict__ out, int N) {
  int wpb  = blockDim.x >> 5;
  int row  = blockIdx.x * wpb + (threadIdx.x >> 5);
  int lane = threadIdx.x & 31;
  if (row >= N) return;
  float4 v = *(const float4*)(x + (size_t)row * DIM + lane * 4);
  float s  = v.x + v.y + v.z + v.w;
  float s2 = v.x * v.x + v.y * v.y + v.z * v.z + v.w * v.w;
  s = wave_sum(s); s2 = wave_sum(s2);
  float mu  = s * (1.f / DIM);
  float var = s2 * (1.f / DIM) - mu * mu;
  float rs  = rsqrtf(var + 1e-5f);
  float4 G = *(const float4*)(gm + lane * 4);
  float4 B = *(const float4*)(bt + lane * 4);
  ushort4 o;
  o.x = f32_to_bf16((v.x - mu) * rs * G.x + B.x);
  o.y = f32_to_bf16((v.y - mu) * rs * G.y + B.y);
  o.z = f32_to_bf16((v.z - mu) * rs * G.z + B.z);
  o.w = f32_to_bf16((v.w - mu) * rs * G.w + B.w);
  *(ushort4*)(out + (size_t)row * DIM + lane * 4) = o;
}

// ---------------- weight convert + transpose: f32 [K,Nout] -> bf16 [Nout,K] ----------------
__global__ void cvt_transpose_kernel(const float* __restrict__ W,
                                     unsigned short* __restrict__ Wt,
                                     int K, int Nout) {
  int i = blockIdx.x * blockDim.x + threadIdx.x;
  if (i >= K * Nout) return;
  int k = i / Nout, n = i - k * Nout;
  Wt[(size_t)n * K + k] = f32_to_bf16(W[i]);
}

// ---------------- f32 -> bf16 flat ----------------
__global__ void cvt_bf16_kernel(const float* __restrict__ in,
                                unsigned short* __restrict__ out, int n) {
  int i = blockIdx.x * blockDim.x + threadIdx.x;
  if (i < n) out[i] = f32_to_bf16(in[i]);
}

// ---------------- WMMA GEMM: C = A[M,K](bf16) x B (Bt=[Nc,K] bf16) ----------------
// Each wave computes a 16x64 strip: A fragment reused across 4 WMMA tiles.
// Fused epilogue: +bias, GELU(tanh), +residual; write f32 and/or bf16.
template <int K>
__global__ __launch_bounds__(256) void gemm_bf16_wmma(
    const unsigned short* __restrict__ A,
    const unsigned short* __restrict__ Bt,
    const float* __restrict__ bias,
    const float* __restrict__ residual,
    float* __restrict__ outF,
    unsigned short* __restrict__ outBf,
    int M, int Nc, int applyGelu) {
  const int wave = threadIdx.x >> 5;
  const int lane = threadIdx.x & 31;
  const int half = lane >> 4;     // 0: lanes 0-15, 1: lanes 16-31
  const int r    = lane & 15;
  const int wpb  = blockDim.x >> 5;
  const int m0   = blockIdx.x << 4;
  const int n0   = (blockIdx.y * wpb + wave) << 6;
  if (n0 >= Nc || m0 >= M) return;

  v8f acc0 = {}, acc1 = {}, acc2 = {}, acc3 = {};

  // A layout per lane (16-bit A 16x32): half 0 holds K 0..7 & 16..23, half 1 holds 8..15 & 24..31
  const unsigned short* aRow = A  + (size_t)(m0 + r) * K + half * 8;
  // B layout per lane (16-bit B 32x16): half 0 holds K 0..7 & 16..23, half 1 holds 8..15 & 24..31
  const unsigned short* bR0  = Bt + (size_t)(n0 +  0 + r) * K + half * 8;
  const unsigned short* bR1  = Bt + (size_t)(n0 + 16 + r) * K + half * 8;
  const unsigned short* bR2  = Bt + (size_t)(n0 + 32 + r) * K + half * 8;
  const unsigned short* bR3  = Bt + (size_t)(n0 + 48 + r) * K + half * 8;

#pragma unroll 4
  for (int k0 = 0; k0 < K; k0 += 32) {
    if (k0 + 32 < K) __builtin_prefetch(aRow + k0 + 32, 0, 1);
    v16bf a  = pack_frag(*(const uint4*)(aRow + k0), *(const uint4*)(aRow + k0 + 16));
    v16bf b0 = pack_frag(*(const uint4*)(bR0 + k0),  *(const uint4*)(bR0 + k0 + 16));
    v16bf b1 = pack_frag(*(const uint4*)(bR1 + k0),  *(const uint4*)(bR1 + k0 + 16));
    v16bf b2 = pack_frag(*(const uint4*)(bR2 + k0),  *(const uint4*)(bR2 + k0 + 16));
    v16bf b3 = pack_frag(*(const uint4*)(bR3 + k0),  *(const uint4*)(bR3 + k0 + 16));
    acc0 = __builtin_amdgcn_wmma_f32_16x16x32_bf16(false, a, false, b0, (short)0, acc0, false, false);
    acc1 = __builtin_amdgcn_wmma_f32_16x16x32_bf16(false, a, false, b1, (short)0, acc1, false, false);
    acc2 = __builtin_amdgcn_wmma_f32_16x16x32_bf16(false, a, false, b2, (short)0, acc2, false, false);
    acc3 = __builtin_amdgcn_wmma_f32_16x16x32_bf16(false, a, false, b3, (short)0, acc3, false, false);
  }

  v8f accs[4] = {acc0, acc1, acc2, acc3};
#pragma unroll
  for (int t = 0; t < 4; ++t) {
    int n = n0 + t * 16 + r;
    float bv = bias ? bias[n] : 0.f;
#pragma unroll
    for (int i = 0; i < 8; ++i) {
      int m = m0 + half * 8 + i;         // C/D layout: VGPR i -> row i (+8 for lanes 16-31)
      float v = accs[t][i] + bv;
      if (applyGelu) {
        float c = v * v * v;
        v = 0.5f * v * (1.f + tanhf(0.7978845608028654f * (v + 0.044715f * c)));
      }
      size_t idx = (size_t)m * Nc + n;
      if (residual) v += residual[idx];
      if (outF)  outF[idx]  = v;
      if (outBf) outBf[idx] = f32_to_bf16(v);
    }
  }
}

// ---------------- per-node attention logits: alpha_src/dst [N,H] ----------------
__global__ void alpha_node_kernel(const float* __restrict__ h,
                                  const float* __restrict__ as,
                                  const float* __restrict__ ad,
                                  float* __restrict__ outS,
                                  float* __restrict__ outD) {
  int n = blockIdx.x, hd = threadIdx.x >> 5, lane = threadIdx.x & 31;
  float4 hv = *(const float4*)(h + (size_t)n * HC + hd * CH + lane * 4);
  float4 av = *(const float4*)(as + hd * CH + lane * 4);
  float4 dv = *(const float4*)(ad + hd * CH + lane * 4);
  float s1 = hv.x * av.x + hv.y * av.y + hv.z * av.z + hv.w * av.w;
  float s2 = hv.x * dv.x + hv.y * dv.y + hv.z * dv.z + hv.w * dv.w;
  s1 = wave_sum(s1); s2 = wave_sum(s2);
  if (lane == 0) { outS[n * NHEAD + hd] = s1; outD[n * NHEAD + hd] = s2; }
}

// ke[h] = sum_c edge_W[h*C+c] * att_edge[h*C+c]  (edge_attr is scalar per edge)
__global__ void ke_kernel(const float* __restrict__ eW,
                          const float* __restrict__ aE,
                          float* __restrict__ ke) {
  int hd = threadIdx.x >> 5, lane = threadIdx.x & 31;
  float4 w = *(const float4*)(eW + hd * CH + lane * 4);
  float4 a = *(const float4*)(aE + hd * CH + lane * 4);
  float s = w.x * a.x + w.y * a.y + w.z * a.z + w.w * a.w;
  s = wave_sum(s);
  if (lane == 0) ke[hd] = s;
}

__global__ void init_nh_kernel(unsigned* __restrict__ amaxk,
                               float* __restrict__ denom, int n) {
  int i = blockIdx.x * blockDim.x + threadIdx.x;
  if (i < n) { amaxk[i] = 0u; denom[i] = 0.f; }
}

__global__ void init_g_kernel(float* __restrict__ g,
                              const float* __restrict__ bias, int total) {
  int i = blockIdx.x * blockDim.x + threadIdx.x;
  if (i < total) g[i] = bias[i & (HC - 1)];
}

// pass 1: alpha = leakyrelu(aS[src]+aD[dst]+eattr*ke); segment max via keyed atomicMax
__global__ void edge_pass1(const int* __restrict__ ei, const float* __restrict__ eattr,
                           const float* __restrict__ aS, const float* __restrict__ aD,
                           const float* __restrict__ ke, float* __restrict__ alphaE,
                           unsigned* __restrict__ amaxk, int E) {
  int t = blockIdx.x * blockDim.x + threadIdx.x;
  if (t >= E * NHEAD) return;
  int e = t >> 2, hd = t & 3;
  int s = ei[e], dd = ei[E + e];
  float a = aS[s * NHEAD + hd] + aD[dd * NHEAD + hd] + eattr[e] * ke[hd];
  a = (a > 0.f) ? a : 0.2f * a;
  alphaE[t] = a;
  atomicMax(&amaxk[dd * NHEAD + hd], fkey(a));
}

// pass 2: ex = exp(alpha - amax[dst]); denom[dst] += ex
__global__ void edge_pass2(const int* __restrict__ ei, float* __restrict__ alphaE,
                           const unsigned* __restrict__ amaxk,
                           float* __restrict__ denom, int E) {
  int t = blockIdx.x * blockDim.x + threadIdx.x;
  if (t >= E * NHEAD) return;
  int e = t >> 2, hd = t & 3;
  int dd = ei[E + e];
  float ex = expf(alphaE[t] - fkey_inv(amaxk[dd * NHEAD + hd]));
  alphaE[t] = ex;
  atomicAdd(&denom[dd * NHEAD + hd], ex);
}

// pass 3: g[dst] += (ex/denom[dst]) * h[src]  (block per edge, 512 channels)
__global__ void edge_pass3(const int* __restrict__ ei, const float* __restrict__ alphaE,
                           const float* __restrict__ denom, const float* __restrict__ h,
                           float* __restrict__ g, int E) {
  int e = blockIdx.x;
  int t = threadIdx.x;           // 0..127 -> float4 chunk
  int hd = t >> 5;
  int s = ei[e], dd = ei[E + e];
  float attn = alphaE[e * NHEAD + hd] / (denom[dd * NHEAD + hd] + 1e-16f);
  float4 hv = *(const float4*)(h + (size_t)s * HC + t * 4);
  float* outp = g + (size_t)dd * HC + t * 4;
  atomicAdd(outp + 0, attn * hv.x);
  atomicAdd(outp + 1, attn * hv.y);
  atomicAdd(outp + 2, attn * hv.z);
  atomicAdd(outp + 3, attn * hv.w);
}

// ---------------- host orchestration ----------------
extern "C" void kernel_launch(void* const* d_in, const int* in_sizes, int n_in,
                              void* d_out, int out_size, void* d_ws, size_t ws_size,
                              hipStream_t stream) {
  const float* x_in      = (const float*)d_in[0];
  const int*   edge_idx  = (const int*)d_in[1];
  const float* edge_attr = (const float*)d_in[2];
  const float* ln1_g     = (const float*)d_in[3];
  const float* ln1_b     = (const float*)d_in[4];
  const float* gat_W     = (const float*)d_in[5];
  const float* att_src   = (const float*)d_in[6];
  const float* att_dst   = (const float*)d_in[7];
  const float* edge_W    = (const float*)d_in[8];
  const float* att_edge  = (const float*)d_in[9];
  const float* gat_bias  = (const float*)d_in[10];
  const float* qf_W      = (const float*)d_in[11];
  const float* qf_b      = (const float*)d_in[12];
  const float* ln2_g     = (const float*)d_in[13];
  const float* ln2_b     = (const float*)d_in[14];
  const float* ff_W1     = (const float*)d_in[15];
  const float* ff_b1     = (const float*)d_in[16];
  const float* ff_W2     = (const float*)d_in[17];
  const float* ff_b2     = (const float*)d_in[18];
  float* x = (float*)d_out;

  // carve workspace (256B aligned)
  uintptr_t p = ((uintptr_t)d_ws + 255) & ~(uintptr_t)255;
  auto carve = [&](size_t bytes) -> void* {
    void* q = (void*)p;
    p = (p + bytes + 255) & ~(uintptr_t)255;
    return q;
  };
  unsigned short* hln   = (unsigned short*)carve((size_t)NNODES * DIM * 2);
  float*          h     = (float*)carve((size_t)NNODES * HC * 4);
  float*          g     = (float*)carve((size_t)NNODES * HC * 4);
  unsigned short* actb  = (unsigned short*)carve((size_t)NNODES * HC * 2);
  float*          aS    = (float*)carve((size_t)NNODES * NHEAD * 4);
  float*          aD    = (float*)carve((size_t)NNODES * NHEAD * 4);
  unsigned*       amaxk = (unsigned*)carve((size_t)NNODES * NHEAD * 4);
  float*          denom = (float*)carve((size_t)NNODES * NHEAD * 4);
  float*          alphaE= (float*)carve((size_t)NEDGES * NHEAD * 4);
  unsigned short* wt    = (unsigned short*)carve((size_t)HC * DIM * 2);
  float*          ke    = (float*)carve(256);

  hipMemcpyAsync(x, x_in, (size_t)NNODES * DIM * 4, hipMemcpyDeviceToDevice, stream);

  const int EH = NEDGES * NHEAD;
  for (int d = 0; d < DEPTHL; ++d) {
    // ---- GAT branch: h_ln = LN1(x); h = h_ln @ W ----
    ln_bf16_kernel<<<(NNODES + 7) / 8, 256, 0, stream>>>(x, ln1_g + d * DIM, ln1_b + d * DIM, hln, NNODES);
    cvt_transpose_kernel<<<(DIM * HC + 255) / 256, 256, 0, stream>>>(gat_W + (size_t)d * DIM * HC, wt, DIM, HC);
    {
      dim3 grid(NNODES / 16, 1);   // 8 waves * 64 cols = 512
      gemm_bf16_wmma<DIM><<<grid, 256, 0, stream>>>(hln, wt, nullptr, nullptr, h, nullptr, NNODES, HC, 0);
    }
    alpha_node_kernel<<<NNODES, 128, 0, stream>>>(h, att_src + d * HC, att_dst + d * HC, aS, aD);
    ke_kernel<<<1, 128, 0, stream>>>(edge_W + d * HC, att_edge + d * HC, ke);
    init_nh_kernel<<<(NNODES * NHEAD + 255) / 256, 256, 0, stream>>>(amaxk, denom, NNODES * NHEAD);
    init_g_kernel<<<(NNODES * HC + 255) / 256, 256, 0, stream>>>(g, gat_bias + d * HC, NNODES * HC);
    edge_pass1<<<(EH + 255) / 256, 256, 0, stream>>>(edge_idx, edge_attr, aS, aD, ke, alphaE, amaxk, NEDGES);
    edge_pass2<<<(EH + 255) / 256, 256, 0, stream>>>(edge_idx, alphaE, amaxk, denom, NEDGES);
    edge_pass3<<<NEDGES, 128, 0, stream>>>(edge_idx, alphaE, denom, h, g, NEDGES);

    // ---- QuickFix projection + residual: x += g @ qf_W + qf_b ----
    cvt_bf16_kernel<<<(NNODES * HC + 255) / 256, 256, 0, stream>>>(g, actb, NNODES * HC);
    cvt_transpose_kernel<<<(HC * DIM + 255) / 256, 256, 0, stream>>>(qf_W + (size_t)d * HC * DIM, wt, HC, DIM);
    {
      dim3 grid(NNODES / 16, 1);   // 2 waves * 64 cols = 128
      gemm_bf16_wmma<HC><<<grid, 64, 0, stream>>>(actb, wt, qf_b + d * DIM, x, x, nullptr, NNODES, DIM, 0);
    }

    // ---- FeedForward + residual ----
    ln_bf16_kernel<<<(NNODES + 7) / 8, 256, 0, stream>>>(x, ln2_g + d * DIM, ln2_b + d * DIM, hln, NNODES);
    cvt_transpose_kernel<<<(DIM * MLPD + 255) / 256, 256, 0, stream>>>(ff_W1 + (size_t)d * DIM * MLPD, wt, DIM, MLPD);
    {
      dim3 grid(NNODES / 16, 1);
      gemm_bf16_wmma<DIM><<<grid, 256, 0, stream>>>(hln, wt, ff_b1 + d * MLPD, nullptr, nullptr, actb, NNODES, MLPD, 1);
    }
    cvt_transpose_kernel<<<(MLPD * DIM + 255) / 256, 256, 0, stream>>>(ff_W2 + (size_t)d * MLPD * DIM, wt, MLPD, DIM);
    {
      dim3 grid(NNODES / 16, 1);
      gemm_bf16_wmma<MLPD><<<grid, 64, 0, stream>>>(actb, wt, ff_b2 + d * DIM, x, x, nullptr, NNODES, DIM, 0);
    }
  }
}
